// S4sequence_68427418960361
// MI455X (gfx1250) — matrix operand order repaired and checked by reading, not compiled, not run on hardware
//
#include <hip/hip_runtime.h>

// ---------------------------------------------------------------------------
// S4 sequence layer forward (B=8, L=4096, H=128, N=64), fp32 in/out.
//   1) per-batch LayerNorm over (L,H)
//   2) K[H,L] via DPLR Cauchy transfer function at roots of unity
//   3) circular conv per (b,h): FFT(4096) in LDS -> *K -> IFFT
//   4) exact GELU, stored as f16 activations
//   5) [32768,128] @ W1^T (128x128) via v_wmma_f32_16x16x32_f16 + bias + skip
// ---------------------------------------------------------------------------

#define B_SZ 8
#define LSEQ 4096
#define HDIM 128
#define NSTATE 64
#define MROWS (B_SZ * LSEQ)       // 32768
#define STEP_F 0.001f
#define EPS_F 1e-5f
#define TWO_PI 6.283185307179586f

typedef __attribute__((ext_vector_type(16))) _Float16 v16h;
typedef __attribute__((ext_vector_type(8)))  _Float16 v8h;
typedef __attribute__((ext_vector_type(8)))  float    v8f;

// ---------------------------- complex helpers ------------------------------
struct cpx { float r, i; };
__device__ __forceinline__ cpx cmul(cpx a, cpx b) { return { a.r * b.r - a.i * b.i, a.r * b.i + a.i * b.r }; }
__device__ __forceinline__ cpx cadd(cpx a, cpx b) { return { a.r + b.r, a.i + b.i }; }
__device__ __forceinline__ cpx csub(cpx a, cpx b) { return { a.r - b.r, a.i - b.i }; }
__device__ __forceinline__ cpx cdiv(cpx a, cpx b) {
    float d = __frcp_rn(b.r * b.r + b.i * b.i);
    return { (a.r * b.r + a.i * b.i) * d, (a.i * b.r - a.r * b.i) * d };
}

// ------------------- kernel 1: per-batch mean / rstd ------------------------
__global__ __launch_bounds__(256) void s4_stats_kernel(const float* __restrict__ x,
                                                       float* __restrict__ stats) {
    __shared__ float s_sum[256];
    __shared__ float s_sq[256];
    const int b = blockIdx.x;
    const int tid = threadIdx.x;
    const size_t n = (size_t)LSEQ * HDIM;
    const float* xb = x + (size_t)b * n;
    float acc = 0.f, acc2 = 0.f;
    for (size_t i = tid; i < n; i += 256) {
        float v = xb[i];
        acc += v;
        acc2 += v * v;
    }
    s_sum[tid] = acc;
    s_sq[tid] = acc2;
    __syncthreads();
    for (int off = 128; off > 0; off >>= 1) {
        if (tid < off) {
            s_sum[tid] += s_sum[tid + off];
            s_sq[tid]  += s_sq[tid + off];
        }
        __syncthreads();
    }
    if (tid == 0) {
        float inv_n = 1.0f / (float)n;
        float mu = s_sum[0] * inv_n;
        float var = s_sq[0] * inv_n - mu * mu;
        stats[2 * b + 0] = mu;
        stats[2 * b + 1] = rsqrtf(var + EPS_F);
    }
}

// ------------------- kernel 2: W1 -> f16 ------------------------------------
__global__ __launch_bounds__(256) void s4_w1cvt_kernel(const float* __restrict__ w1,
                                                       _Float16* __restrict__ w1h) {
    int i = blockIdx.x * 256 + threadIdx.x;
    if (i < HDIM * HDIM) w1h[i] = (_Float16)w1[i];
}

// ------------------- kernel 3: Cauchy / DPLR transfer function --------------
// grid = H * (L/256); block handles one h and 256 consecutive l values.
__global__ __launch_bounds__(256) void s4_cauchy_kernel(
    const float* __restrict__ LamR, const float* __restrict__ LamI,
    const float* __restrict__ Pr,   const float* __restrict__ Pi,
    const float* __restrict__ Br,   const float* __restrict__ Bi,
    const float* __restrict__ Cr,   const float* __restrict__ Ci,
    float* __restrict__ Kr, float* __restrict__ Ki) {
    __shared__ float s_lr[NSTATE], s_li[NSTATE];
    __shared__ float s00r[NSTATE], s00i[NSTATE];
    __shared__ float s01r[NSTATE], s01i[NSTATE];
    __shared__ float s10r[NSTATE], s10i[NSTATE];
    __shared__ float s11r[NSTATE], s11i[NSTATE];

    const int h = blockIdx.x >> 4;            // L/256 = 16 blocks per h
    const int l0 = (blockIdx.x & 15) * 256;
    const int tid = threadIdx.x;

    if (tid < NSTATE) {
        int o = h * NSTATE + tid;
        float cr = Cr[o], ci = Ci[o];
        float br = Br[o], bi = Bi[o];
        float pr = Pr[o], pi = Pi[o];
        s_lr[tid] = LamR[o];
        s_li[tid] = LamI[o];
        // conj(C)*B
        s00r[tid] = cr * br + ci * bi;  s00i[tid] = cr * bi - ci * br;
        // conj(C)*P
        s01r[tid] = cr * pr + ci * pi;  s01i[tid] = cr * pi - ci * pr;
        // conj(P)*B
        s10r[tid] = pr * br + pi * bi;  s10i[tid] = pr * bi - pi * br;
        // conj(P)*P
        s11r[tid] = pr * pr + pi * pi;  s11i[tid] = 0.0f;
    }
    __syncthreads();

    const int l = l0 + tid;
    float ang = -TWO_PI * (float)l / (float)LSEQ;
    float so, co;
    __sincosf(ang, &so, &co);
    cpx omega = { co, so };
    cpx one_m = { 1.0f - omega.r, -omega.i };
    cpx one_p = { 1.0f + omega.r,  omega.i };
    cpx g  = cdiv(one_m, one_p);
    g.r *= (2.0f / STEP_F);
    g.i *= (2.0f / STEP_F);
    cpx c2 = cdiv({ 2.0f, 0.0f }, one_p);

    cpx k00 = { 0, 0 }, k01 = { 0, 0 }, k10 = { 0, 0 }, k11 = { 0, 0 };
#pragma unroll 4
    for (int n = 0; n < NSTATE; ++n) {
        float dr = g.r - s_lr[n];
        float di = g.i - s_li[n];
        float d = __frcp_rn(dr * dr + di * di);
        cpx inv = { dr * d, -di * d };
        k00 = cadd(k00, cmul({ s00r[n], s00i[n] }, inv));
        k01 = cadd(k01, cmul({ s01r[n], s01i[n] }, inv));
        k10 = cadd(k10, cmul({ s10r[n], s10i[n] }, inv));
        k11 = cadd(k11, cmul({ s11r[n], s11i[n] }, inv));
    }
    cpx den = { 1.0f + k11.r, k11.i };
    cpx kk = csub(k00, cdiv(cmul(k01, k10), den));
    cpx K = cmul(c2, kk);
    Kr[h * LSEQ + l] = K.r;
    Ki[h * LSEQ + l] = K.i;
}

// ------------------- kernel 4: fused norm + FFT conv + GELU -----------------
// One workgroup per (b,h); 4096-pt complex FFT in LDS (32 KB).
__global__ __launch_bounds__(256) void s4_fftconv_kernel(
    const float* __restrict__ x, const float* __restrict__ stats,
    const float* __restrict__ Kr, const float* __restrict__ Ki,
    _Float16* __restrict__ act) {
    __shared__ float2 s[LSEQ];
    const int bh = blockIdx.x;
    const int b = bh >> 7;
    const int h = bh & (HDIM - 1);
    const int tid = threadIdx.x;

    const float mu = stats[2 * b + 0];
    const float rs = stats[2 * b + 1];
    const float* xp = x + (size_t)b * LSEQ * HDIM + h;

    for (int i = tid; i < LSEQ; i += 256) {
        float v = (xp[(size_t)i * HDIM] - mu) * rs;
        s[i] = make_float2(v, 0.0f);
    }
    __syncthreads();

    // forward FFT: decimation-in-frequency, output bit-reversed
    for (int len = LSEQ; len >= 2; len >>= 1) {
        int half = len >> 1;
        for (int i = tid; i < (LSEQ >> 1); i += 256) {
            int j = i & (half - 1);
            int idx1 = ((i & ~(half - 1)) << 1) | j;
            int idx2 = idx1 + half;
            float sw, cw;
            __sincosf(-TWO_PI * (float)j / (float)len, &sw, &cw);
            float2 a = s[idx1], c = s[idx2];
            s[idx1] = make_float2(a.x + c.x, a.y + c.y);
            float tr = a.x - c.x, ti = a.y - c.y;
            s[idx2] = make_float2(tr * cw - ti * sw, tr * sw + ti * cw);
        }
        __syncthreads();
    }

    // pointwise multiply by K (frequency index = bit-reverse of LDS index)
    const float* krh = Kr + (size_t)h * LSEQ;
    const float* kih = Ki + (size_t)h * LSEQ;
    for (int i = tid; i < LSEQ; i += 256) {
        int f = (int)(__brev((unsigned)i) >> 20);   // 12-bit reversal
        float kr = krh[f], ki = kih[f];
        float2 v = s[i];
        s[i] = make_float2(v.x * kr - v.y * ki, v.x * ki + v.y * kr);
    }
    __syncthreads();

    // inverse FFT: decimation-in-time, consumes bit-reversed, natural output
    for (int len = 2; len <= LSEQ; len <<= 1) {
        int half = len >> 1;
        for (int i = tid; i < (LSEQ >> 1); i += 256) {
            int j = i & (half - 1);
            int idx1 = ((i & ~(half - 1)) << 1) | j;
            int idx2 = idx1 + half;
            float sw, cw;
            __sincosf(TWO_PI * (float)j / (float)len, &sw, &cw);
            float2 a = s[idx1], c = s[idx2];
            float tr = c.x * cw - c.y * sw;
            float ti = c.x * sw + c.y * cw;
            s[idx1] = make_float2(a.x + tr, a.y + ti);
            s[idx2] = make_float2(a.x - tr, a.y - ti);
        }
        __syncthreads();
    }

    // scale 1/L, exact GELU, store f16 activation [B*L, H]
    const float inv_l = 1.0f / (float)LSEQ;
    const float inv_sqrt2 = 0.7071067811865476f;
    for (int i = tid; i < LSEQ; i += 256) {
        float v = s[i].x * inv_l;
        float gel = 0.5f * v * (1.0f + erff(v * inv_sqrt2));
        act[((size_t)(b * LSEQ + i)) * HDIM + h] = (_Float16)gel;
    }
}

// ------------------- kernel 5: WMMA GEMM + bias + residual ------------------
// out[m, n] = sum_k act[m,k] * W1[n,k] + b1[n] + skip[m,n]
// grid = (M/128, H/16); block = 256 threads = 8 waves, one 16x16 tile per wave.
__global__ __launch_bounds__(256) void s4_gemm_kernel(
    const _Float16* __restrict__ act, const _Float16* __restrict__ w1h,
    const float* __restrict__ b1, const float* __restrict__ skip,
    float* __restrict__ out) {
    const int lane = threadIdx.x & 31;
    const int wave = threadIdx.x >> 5;
    const int m0 = blockIdx.x * 128 + wave * 16;
    const int n0 = blockIdx.y * 16;
    const int l15 = lane & 15;
    const int hsel = lane >> 4;      // 0 for lanes 0-15, 1 for lanes 16-31

    // A fragment (16x32 f16): lane holds row m0+l15; elements are 16
    // consecutive k values starting at k0 + 16*hsel.
    const _Float16* rowA = act + (size_t)(m0 + l15) * HDIM;
    // B fragment (32x16 f16): lane holds column n0+l15; elements are 16
    // consecutive k values starting at k0 + 16*hsel.
    const _Float16* rowB = w1h + (size_t)(n0 + l15) * HDIM + hsel * 16;

    v8f c = {};
#pragma unroll
    for (int k0 = 0; k0 < HDIM; k0 += 32) {
        v16h a = *(const v16h*)(rowA + k0 + 16 * hsel);
        v16h bm = *(const v16h*)(rowB + k0);
        c = __builtin_amdgcn_wmma_f32_16x16x32_f16(
                /*neg_a=*/false, a, /*neg_b=*/false, bm,
                /*c_mod=*/(short)0, c, /*reuse_a=*/false, /*reuse_b=*/false);
    }

    const float bias = b1[n0 + l15];
#pragma unroll
    for (int r = 0; r < 8; ++r) {
        int row = m0 + r + 8 * hsel;      // C/D layout: VGPR r -> M = r (+8)
        size_t idx = (size_t)row * HDIM + n0 + l15;
        out[idx] = c[r] + bias + skip[idx];
    }
}

// ---------------------------------------------------------------------------
extern "C" void kernel_launch(void* const* d_in, const int* in_sizes, int n_in,
                              void* d_out, int out_size, void* d_ws, size_t ws_size,
                              hipStream_t stream) {
    (void)in_sizes; (void)n_in; (void)out_size; (void)ws_size;
    const float* x    = (const float*)d_in[0];
    const float* LamR = (const float*)d_in[1];
    const float* LamI = (const float*)d_in[2];
    const float* Pr   = (const float*)d_in[3];
    const float* Pi   = (const float*)d_in[4];
    const float* Br   = (const float*)d_in[5];
    const float* Bi   = (const float*)d_in[6];
    const float* Cr   = (const float*)d_in[7];
    const float* Ci   = (const float*)d_in[8];
    const float* W1   = (const float*)d_in[9];
    const float* b1   = (const float*)d_in[10];
    float* out = (float*)d_out;

    // workspace carve (all offsets 256B-aligned)
    char* base = (char*)d_ws;
    float*    stats = (float*)base;                                   //   64 B
    float*    Kr    = (float*)(base + 256);                           //  2 MB
    float*    Ki    = Kr + (size_t)HDIM * LSEQ;                       //  2 MB
    _Float16* W1h   = (_Float16*)(base + 256 + 2ull * HDIM * LSEQ * 4); // 32 KB
    _Float16* act   = W1h + (size_t)HDIM * HDIM;                      //  8 MB

    s4_stats_kernel<<<B_SZ, 256, 0, stream>>>(x, stats);
    s4_w1cvt_kernel<<<(HDIM * HDIM + 255) / 256, 256, 0, stream>>>(W1, W1h);
    s4_cauchy_kernel<<<HDIM * (LSEQ / 256), 256, 0, stream>>>(
        LamR, LamI, Pr, Pi, Br, Bi, Cr, Ci, Kr, Ki);
    s4_fftconv_kernel<<<B_SZ * HDIM, 256, 0, stream>>>(x, stats, Kr, Ki, act);
    s4_gemm_kernel<<<dim3(MROWS / 128, HDIM / 16), 256, 0, stream>>>(
        act, W1h, b1, x, out);
}
